// SkeletonMotionQuantizer_34660386079096
// MI455X (gfx1250) — matrix operand order repaired and checked by reading, not compiled, not run on hardware
//
#include <hip/hip_runtime.h>
#include <stdint.h>

// ---------- vector typedefs ----------
typedef _Float16 v16h __attribute__((ext_vector_type(16)));
typedef _Float16 v4h  __attribute__((ext_vector_type(4)));
typedef float    v8f  __attribute__((ext_vector_type(8)));
typedef float    v4f  __attribute__((ext_vector_type(4)));
typedef unsigned int v4u __attribute__((ext_vector_type(4)));
typedef int      v8i  __attribute__((ext_vector_type(8)));
typedef int      v4i  __attribute__((ext_vector_type(4)));

// ---------- problem constants ----------
// B=32, T=512, D=128, K=1024, W=8 -> P=64, N=2048 patches
#define NPATCH 2048
#define KCODES 1024
#define WDIM   8
#define DDIM   128
#define PATCH_ELEMS 1024   // W*D

// output layout (floats), return order: quantize, encoding_indices, loss, distances
#define OUT_Q    ((size_t)0)
#define OUT_ENC  ((size_t)2097152)            // 32*512*128
#define OUT_LOSS ((size_t)2113536)            // + 32*512
#define OUT_DIST ((size_t)2113537)            // + 1

// workspace layout (bytes)
#define WS_EHI   ((size_t)0)                  // 1024*1024 halves = 2 MB
#define WS_ELO   ((size_t)2097152)
#define WS_E2    ((size_t)4194304)            // 1024*8 f32
#define WS_VALID ((size_t)4227072)            // 2048 f32
#define WS_CVAL  ((size_t)4235264)            // 2048*16 f32
#define WS_CIDX  ((size_t)4366336)            // 2048*16 i32
#define WS_IDX   ((size_t)4497408)            // 2048 i32
#define WS_PART  ((size_t)4505600)            // 256*2 f32

// LDS layout (bytes) inside qdist_main
#define L_AHI   0        // [32][136] halves
#define L_ALO   8704
#define L_BHI   17408    // [64][136] halves
#define L_BLO   34816
#define L_X2S   52224    // [32][8] f32
#define L_TOTAL 53248
#define ASTRIDE 136      // halves per A row (128 + 8 pad)
#define BSTRIDE 136      // halves per B row
#define DSTRIDE 68       // floats per distS row (overlay @ L_BHI)

// ---------- helpers ----------
__device__ __forceinline__ float wred_sum(float v) {
#pragma unroll
  for (int o = 16; o > 0; o >>= 1) v += __shfl_xor(v, o, 32);
  return v;
}

// Tensor Data Mover: 2D tile load, 64 rows x 128 halves, row stride 1024 halves,
// LDS destination padded 8 halves (4 DWORDs) after every 256B row -> 136-half stride.
__device__ __forceinline__ void tdm_load_tile64x128(uint32_t lds_off, uint64_t gaddr) {
  v4u g0;
  v8i g1;
  v4i g2 = {0, 0, 0, 0};
  v4i g3 = {0, 0, 0, 0};
  v8i g4 = {0, 0, 0, 0, 0, 0, 0, 0};
  // group0: count=1 | lds_addr | global_addr[31:0] | global_addr[56:32], type=2
  g0[0] = 1u;
  g0[1] = lds_off;
  g0[2] = (uint32_t)gaddr;
  g0[3] = ((uint32_t)(gaddr >> 32)) | (2u << 30);
  // group1: data_size=1(2B), pad_enable, pad_interval=5 (64 DW = 256B), pad_amount=3 (4 DW = 16B)
  g1[0] = (int)((1u << 16) | (1u << 20) | (5u << 22) | (3u << 25));
  g1[1] = (int)(128u << 16);   // tensor_dim0[15:0] = 128
  g1[2] = (int)(64u << 16);    // tensor_dim0 hi=0 | tensor_dim1[15:0] = 64
  g1[3] = (int)(128u << 16);   // tensor_dim1 hi=0 | tile_dim0 = 128
  g1[4] = 64;                  // tile_dim1 = 64 | tile_dim2 = 0
  g1[5] = 1024;                // tensor_dim0_stride lo32 = 1024 elems
  g1[6] = 0;
  g1[7] = 0;
  __builtin_amdgcn_tensor_load_to_lds(g0, g1, g2, g3, g4, 0);
}

union Frag { v16h v; v4u q[2]; };

__device__ __forceinline__ v16h loadA(const _Float16* p) {
  Frag u;
  u.q[0] = *(const v4u*)(p);       // K segment 0..7
  u.q[1] = *(const v4u*)(p + 16);  // K segment 16..23
  return u.v;
}
__device__ __forceinline__ v16h loadB(const _Float16* p) {
  Frag u;
  u.q[0] = *(const v4u*)(p);       // K segment 0..7
  u.q[1] = *(const v4u*)(p + 16);  // K segment 16..23
  return u.v;
}

#define WMMA_F16(A, B, C) \
  __builtin_amdgcn_wmma_f32_16x16x32_f16(false, (A), false, (B), (short)0, (C), false, false)

// ---------- kernel 1: split embedding to f16 hi/lo + e2 ----------
__global__ __launch_bounds__(256) void prep_embed(const float* __restrict__ emb,
                                                  _Float16* __restrict__ eHi,
                                                  _Float16* __restrict__ eLo,
                                                  float* __restrict__ e2) {
  int k = blockIdx.x;                // 0..1023
  int wv = threadIdx.x >> 5;         // w slice 0..7
  int ln = threadIdx.x & 31;
  size_t base = (size_t)k * PATCH_ELEMS + (size_t)wv * DDIM;
  v4f x = ((const v4f*)(emb + base))[ln];
  v4h hi, lo;
  float s = 0.f;
#pragma unroll
  for (int c = 0; c < 4; ++c) {
    float f = x[c];
    _Float16 h = (_Float16)f;
    hi[c] = h;
    lo[c] = (_Float16)(f - (float)h);
    s = fmaf(f, f, s);
  }
  *(v4h*)(eHi + base + ln * 4) = hi;
  *(v4h*)(eLo + base + ln * 4) = lo;
  s = wred_sum(s);
  if (ln == 0) e2[k * 8 + wv] = s;
}

// ---------- kernel 2: per-patch validity from mask ----------
__global__ __launch_bounds__(256) void prep_valid(const float* __restrict__ mask,
                                                  float* __restrict__ validf) {
  int n = blockIdx.x * 8 + (threadIdx.x >> 5);
  int ln = threadIdx.x & 31;
  const v4f* m = (const v4f*)(mask + (size_t)n * PATCH_ELEMS);
  float s = 0.f;
#pragma unroll
  for (int i = 0; i < 8; ++i) {
    v4f v = m[ln + 32 * i];
    s += v[0] + v[1] + v[2] + v[3];
  }
  s = wred_sum(s);
  if (ln == 0) validf[n] = (s > 0.f) ? 1.f : 0.f;
}

// ---------- kernel 3: main distance GEMM (WMMA + TDM) ----------
__global__ __launch_bounds__(256) void qdist_main(const float* __restrict__ x,
                                                  const _Float16* __restrict__ eHi,
                                                  const _Float16* __restrict__ eLo,
                                                  const float* __restrict__ e2,
                                                  const float* __restrict__ validf,
                                                  float* __restrict__ outDist,
                                                  float* __restrict__ candVal,
                                                  int* __restrict__ candIdx) {
  __shared__ __align__(16) unsigned char smem[L_TOTAL];
  _Float16* aHi = (_Float16*)(smem + L_AHI);
  _Float16* aLo = (_Float16*)(smem + L_ALO);
  _Float16* bHi = (_Float16*)(smem + L_BHI);
  _Float16* bLo = (_Float16*)(smem + L_BLO);
  float* x2s = (float*)(smem + L_X2S);
  float* distS = (float*)(smem + L_BHI);   // overlay after compute

  const int n0 = blockIdx.x * 32;          // patch tile base
  const int kb = blockIdx.y;               // k-block 0..15
  const int k0 = kb * 64;                  // code tile base
  const int wv = threadIdx.x >> 5;
  const int ln = threadIdx.x & 31;
  const int sr = wv >> 2;                  // subtile row 0..1
  const int sc = wv & 3;                   // subtile col 0..3
  const int hl = (ln >> 4) & 1;            // half-wave

  const uint32_t ldsBHi = (uint32_t)(uintptr_t)bHi;  // low 32 bits of flat LDS ptr = LDS offset
  const uint32_t ldsBLo = (uint32_t)(uintptr_t)bLo;

  v8f acc[8];
#pragma unroll
  for (int w = 0; w < 8; ++w) acc[w] = (v8f){0.f, 0.f, 0.f, 0.f, 0.f, 0.f, 0.f, 0.f};

  const int arow = sr * 16 + (ln & 15);
  const int bcol = sc * 16 + (ln & 15);

#pragma unroll
  for (int w = 0; w < WDIM; ++w) {
    __syncthreads();  // previous iteration's LDS fully consumed
    if (wv == 0) {
      uint64_t off = ((uint64_t)k0 * PATCH_ELEMS + (uint64_t)w * DDIM) * 2;
      tdm_load_tile64x128(ldsBHi, (uint64_t)(uintptr_t)eHi + off);
      tdm_load_tile64x128(ldsBLo, (uint64_t)(uintptr_t)eLo + off);
    }
    // convert A rows (4 per wave) + x2 reduction
#pragma unroll
    for (int j = 0; j < 4; ++j) {
      int r = wv * 4 + j;
      v4f xv = ((const v4f*)(x + (size_t)(n0 + r) * PATCH_ELEMS + (size_t)w * DDIM))[ln];
      v4h hi, lo;
      float s = 0.f;
#pragma unroll
      for (int c = 0; c < 4; ++c) {
        float f = xv[c];
        _Float16 h = (_Float16)f;
        hi[c] = h;
        lo[c] = (_Float16)(f - (float)h);
        s = fmaf(f, f, s);
      }
      *(v4h*)(aHi + r * ASTRIDE + ln * 4) = hi;
      *(v4h*)(aLo + r * ASTRIDE + ln * 4) = lo;
      s = wred_sum(s);
      if (ln == 0) x2s[r * 8 + w] = s;
    }
    if (wv == 0) __builtin_amdgcn_s_wait_tensorcnt(0);
    __syncthreads();  // TDM data + A conversions visible

#pragma unroll
    for (int kc = 0; kc < 4; ++kc) {
      const _Float16* pa = aHi + arow * ASTRIDE + kc * 32 + hl * 8;
      const _Float16* pl = aLo + arow * ASTRIDE + kc * 32 + hl * 8;
      const _Float16* pbh = bHi + bcol * BSTRIDE + kc * 32 + hl * 8;
      const _Float16* pbl = bLo + bcol * BSTRIDE + kc * 32 + hl * 8;
      v16h AH = loadA(pa);
      v16h AL = loadA(pl);
      v16h BH = loadB(pbh);
      v16h BL = loadB(pbl);
      acc[w] = WMMA_F16(AH, BH, acc[w]);  // hi*hi
      acc[w] = WMMA_F16(AH, BL, acc[w]);  // hi*lo
      acc[w] = WMMA_F16(AL, BH, acc[w]);  // lo*hi
    }
  }

  // ---- epilogue: distances ----
  const int colk = k0 + sc * 16 + (ln & 15);
  v4f e2a = *(const v4f*)(e2 + colk * 8);
  v4f e2b = *(const v4f*)(e2 + colk * 8 + 4);
  float e2l[8] = {e2a[0], e2a[1], e2a[2], e2a[3], e2b[0], e2b[1], e2b[2], e2b[3]};

  __syncthreads();  // all frag reads of b-buffer done -> safe to overlay distS

#pragma unroll
  for (int r = 0; r < 8; ++r) {
    int row = sr * 16 + r + hl * 8;  // 0..31
    float vf = validf[n0 + row];
    float dsum = 0.f;
#pragma unroll
    for (int w = 0; w < 8; ++w) {
      float xe = acc[w][r];
      float sq = x2s[row * 8 + w] + e2l[w] - 2.f * xe;
      dsum += __builtin_sqrtf(fmaxf(sq, 0.f));
    }
    distS[row * DSTRIDE + sc * 16 + (ln & 15)] = -dsum * vf;  // where(valid, dist, 0)
  }
  __syncthreads();

  // write all W repeated distance rows, coalesced
  for (int rr = wv; rr < 256; rr += 8) {
    int r = rr >> 3, wt = rr & 7;
    int n = n0 + r, b = n >> 6, p = n & 63;
    size_t base = ((size_t)(b * 512 + p * 8 + wt)) * (size_t)KCODES + (size_t)k0;
    outDist[base + ln * 2] = distS[r * DSTRIDE + ln * 2];
    outDist[base + ln * 2 + 1] = distS[r * DSTRIDE + ln * 2 + 1];
  }

  // per-patch argmax candidates over this 64-wide k-block (first index on ties)
#pragma unroll
  for (int j = 0; j < 4; ++j) {
    int r = wv * 4 + j;
    float v0 = distS[r * DSTRIDE + ln * 2];
    float v1 = distS[r * DSTRIDE + ln * 2 + 1];
    float bv = v0;
    int bi = ln * 2;
    if (v1 > bv) { bv = v1; bi = ln * 2 + 1; }
#pragma unroll
    for (int o = 16; o > 0; o >>= 1) {
      float ov = __shfl_xor(bv, o, 32);
      int oi = __shfl_xor(bi, o, 32);
      if (ov > bv || (ov == bv && oi < bi)) { bv = ov; bi = oi; }
    }
    if (ln == 0) {
      candVal[(size_t)(n0 + r) * 16 + kb] = bv;
      candIdx[(size_t)(n0 + r) * 16 + kb] = k0 + bi;
    }
  }
}

// ---------- kernel 4: reduce candidates -> idx, encoding_indices ----------
__global__ __launch_bounds__(256) void finalize_idx(const float* __restrict__ candVal,
                                                    const int* __restrict__ candIdx,
                                                    const float* __restrict__ validf,
                                                    int* __restrict__ idxN,
                                                    float* __restrict__ outEnc) {
  int n = blockIdx.x * 256 + threadIdx.x;
  if (n >= NPATCH) return;
  float bv = candVal[(size_t)n * 16];
  int bi = candIdx[(size_t)n * 16];
#pragma unroll
  for (int j = 1; j < 16; ++j) {
    float v = candVal[(size_t)n * 16 + j];
    int i = candIdx[(size_t)n * 16 + j];
    if (v > bv || (v == bv && i < bi)) { bv = v; bi = i; }
  }
  int pidx = (validf[n] > 0.f) ? bi : 0;
  idxN[n] = pidx;
  int b = n >> 6, p = n & 63;
#pragma unroll
  for (int wt = 0; wt < 8; ++wt)
    outEnc[(size_t)b * 512 + p * 8 + wt] = (float)pidx;
}

// ---------- kernel 5: gather quantize + loss partials ----------
__global__ __launch_bounds__(256) void quant_loss(const float* __restrict__ emb,
                                                  const int* __restrict__ idxN,
                                                  const float* __restrict__ validf,
                                                  const float* __restrict__ x,
                                                  const float* __restrict__ mask,
                                                  float* __restrict__ outQ,
                                                  float* __restrict__ partials) {
  __shared__ float r1[8], r2[8];
  int wv = threadIdx.x >> 5, ln = threadIdx.x & 31;
  int n = blockIdx.x * 8 + wv;
  int idx = idxN[n];
  float vf = validf[n];
  const v4f* pe = (const v4f*)(emb + (size_t)idx * PATCH_ELEMS);
  const v4f* px = (const v4f*)(x + (size_t)n * PATCH_ELEMS);
  const v4f* pm = (const v4f*)(mask + (size_t)n * PATCH_ELEMS);
  v4f* pq = (v4f*)(outQ + (size_t)n * PATCH_ELEMS);
  float s1 = 0.f, s2 = 0.f;
#pragma unroll
  for (int i = 0; i < 8; ++i) {
    int t = ln + 32 * i;
    v4f e = pe[t], xv = px[t], m = pm[t];
    v4f q;
#pragma unroll
    for (int c = 0; c < 4; ++c) q[c] = vf * e[c];
    pq[t] = q;
#pragma unroll
    for (int c = 0; c < 4; ++c) {
      float d = q[c] - xv[c];
      s1 = fmaf(d * d, m[c], s1);
      s2 += m[c];
    }
  }
  s1 = wred_sum(s1);
  s2 = wred_sum(s2);
  if (ln == 0) { r1[wv] = s1; r2[wv] = s2; }
  __syncthreads();
  if (threadIdx.x == 0) {
    float a = 0.f, b = 0.f;
#pragma unroll
    for (int i = 0; i < 8; ++i) { a += r1[i]; b += r2[i]; }
    partials[blockIdx.x * 2] = a;
    partials[blockIdx.x * 2 + 1] = b;
  }
}

// ---------- kernel 6: final loss ----------
__global__ __launch_bounds__(256) void loss_final(const float* __restrict__ partials,
                                                  float* __restrict__ outLoss) {
  __shared__ float s1[256], s2[256];
  int t = threadIdx.x;
  s1[t] = partials[t * 2];
  s2[t] = partials[t * 2 + 1];
  __syncthreads();
  for (int o = 128; o > 0; o >>= 1) {
    if (t < o) { s1[t] += s1[t + o]; s2[t] += s2[t + o]; }
    __syncthreads();
  }
  if (t == 0) outLoss[0] = 0.25f * s1[0] / s2[0];
}

// ---------- host launcher ----------
extern "C" void kernel_launch(void* const* d_in, const int* in_sizes, int n_in,
                              void* d_out, int out_size, void* d_ws, size_t ws_size,
                              hipStream_t stream) {
  (void)in_sizes; (void)n_in; (void)out_size; (void)ws_size;
  const float* x = (const float*)d_in[0];
  const float* mask = (const float*)d_in[1];
  const float* emb = (const float*)d_in[2];
  float* out = (float*)d_out;
  unsigned char* ws = (unsigned char*)d_ws;

  _Float16* eHi = (_Float16*)(ws + WS_EHI);
  _Float16* eLo = (_Float16*)(ws + WS_ELO);
  float* e2 = (float*)(ws + WS_E2);
  float* validf = (float*)(ws + WS_VALID);
  float* candVal = (float*)(ws + WS_CVAL);
  int* candIdx = (int*)(ws + WS_CIDX);
  int* idxN = (int*)(ws + WS_IDX);
  float* partials = (float*)(ws + WS_PART);

  prep_embed<<<KCODES, 256, 0, stream>>>(emb, eHi, eLo, e2);
  prep_valid<<<NPATCH / 8, 256, 0, stream>>>(mask, validf);
  qdist_main<<<dim3(NPATCH / 32, KCODES / 64), 256, 0, stream>>>(
      x, eHi, eLo, e2, validf, out + OUT_DIST, candVal, candIdx);
  finalize_idx<<<NPATCH / 256, 256, 0, stream>>>(candVal, candIdx, validf, idxN, out + OUT_ENC);
  quant_loss<<<NPATCH / 8, 256, 0, stream>>>(emb, idxN, validf, x, mask, out + OUT_Q, partials);
  loss_final<<<1, 256, 0, stream>>>(partials, out + OUT_LOSS);
}
